// AtomInteractionWithResidual_79396765433907
// MI455X (gfx1250) — matrix operand order repaired and checked by reading, not compiled, not run on hardware
//
#include <hip/hip_runtime.h>

// ---------------------------------------------------------------------------
// PhysNet interaction block on gfx1250 (CDNA5, wave32, WMMA f16 -> f32 acc).
//  - rbf filter GEMM fused with neighbor gather (g never touches HBM)
//  - entire 12-layer per-atom MLP chain fused in one kernel, state in LDS
//  - weight staging uses GLOBAL_LOAD_ASYNC_TO_LDS_B128 when available
// ---------------------------------------------------------------------------

typedef _Float16 v16h __attribute__((ext_vector_type(16)));
typedef float    v8f  __attribute__((ext_vector_type(8)));
typedef int      v4i  __attribute__((ext_vector_type(4)));

#define BD 32
#define ND 256
#define MD 64
#define FD 128
#define KD 64
#define ROWS (BD * ND)          // 8192 atom rows

#if defined(__AMDGCN__) && __has_builtin(__builtin_amdgcn_global_load_async_to_lds_b128) && __has_builtin(__builtin_amdgcn_s_wait_asynccnt)
#define USE_ASYNC_LDS 1
#else
#define USE_ASYNC_LDS 0
#endif

#define AS1 __attribute__((address_space(1)))
#define AS3 __attribute__((address_space(3)))

// shifted softplus: log(1+exp(x)) - log(2), numerically stable
__device__ __forceinline__ float sspf(float x) {
    return fmaxf(x, 0.0f) + __logf(1.0f + __expf(-fabsf(x))) - 0.69314718055994531f;
}

// Stage `chunks` 16-byte chunks from global -> LDS with all 256 threads.
__device__ __forceinline__ void stage16(const void* __restrict__ g,
                                        void* __restrict__ s, int chunks, int tid) {
    const uint4* gp = (const uint4*)g;
    uint4* sp = (uint4*)s;
#if USE_ASYNC_LDS
    for (int i = tid; i < chunks; i += 256) {
        __builtin_amdgcn_global_load_async_to_lds_b128(
            (AS1 v4i*)(gp + i), (AS3 v4i*)(sp + i), 0, 0);
    }
    __builtin_amdgcn_s_wait_asynccnt(0);
#else
    for (int i = tid; i < chunks; i += 256) sp[i] = gp[i];
#endif
}

// ---- WMMA fragment builders (layouts per CDNA5 ISA 7.12.2) -----------------
// A (16x32 f16): lane L holds row M=L%16; halves 0..7 -> K = kb..kb+7,
// halves 8..15 -> K = kb+16..kb+23, kb = k0 + 8*(L>=16).
__device__ __forceinline__ v16h load_a_frag(const _Float16* base, int ldk,
                                            int row0, int k0, int lane) {
    const _Float16* p = base + (row0 + (lane & 15)) * ldk + k0 + ((lane >> 4) & 1) * 8;
    v16h a;
#pragma unroll
    for (int j = 0; j < 8; ++j) a[j] = p[j];
#pragma unroll
    for (int j = 0; j < 8; ++j) a[8 + j] = p[16 + j];
    return a;
}

// B (32x16 f16) built from W stored [F][K] (B[k][n] = W[f0+n][k]):
// lane L holds col N=L%16; halves 0..7 -> K = kb..kb+7,
// halves 8..15 -> K = kb+16..kb+23, kb = k0 + 8*(L>=16).
__device__ __forceinline__ v16h load_b_frag(const _Float16* w, int ldk,
                                            int f0, int k0, int lane) {
    const _Float16* p = w + (f0 + (lane & 15)) * ldk + k0 + ((lane >> 4) & 1) * 8;
    v16h b;
#pragma unroll
    for (int j = 0; j < 8; ++j) b[j] = p[j];
#pragma unroll
    for (int j = 0; j < 8; ++j) b[8 + j] = p[16 + j];
    return b;
}

// 4 col-tiles (16x16 each) of one row-tile, K = 128 (4 WMMA steps)
__device__ __forceinline__ void gemm4(const _Float16* sH, const _Float16* sW,
                                      int row0, int fbase, int lane, v8f c[4]) {
#pragma unroll
    for (int ct = 0; ct < 4; ++ct) {
        v8f acc = {};
#pragma unroll
        for (int ks = 0; ks < 4; ++ks) {
            v16h a = load_a_frag(sH, FD, row0, ks * 32, lane);
            v16h b = load_b_frag(sW, FD, fbase + ct * 16, ks * 32, lane);
            acc = __builtin_amdgcn_wmma_f32_16x16x32_f16(
                      false, a, false, b, (short)0, acc, false, false);
        }
        c[ct] = acc;
    }
}

// ---- prep kernels ----------------------------------------------------------
__global__ __launch_bounds__(256) void ssp_kernel(const float* __restrict__ x,
                                                  float* __restrict__ y, int n) {
    int i = blockIdx.x * 256 + threadIdx.x;
    if (i < n) y[i] = sspf(x[i]);
}

__global__ __launch_bounds__(256) void cvt_h_kernel(const float* __restrict__ x,
                                                    _Float16* __restrict__ y, int n) {
    int i = blockIdx.x * 256 + threadIdx.x;
    if (i < n) y[i] = (_Float16)x[i];
}

// ---- RBF filter GEMM + neighbor gather-reduce (fused, one block per (b,n)) -
// g[m,f] = sum_k rbf[b,n,m,k] * k2f_W[f,k]   (WMMA, g kept in LDS)
// xj[b,n,f] = sum_m xa[b, nbr[b,n,m], f] * g[m,f]
__global__ __launch_bounds__(256) void rbf_gather_kernel(
        const float* __restrict__ rbf, const int* __restrict__ nbr,
        const float* __restrict__ xa, const _Float16* __restrict__ k2f_h,
        float* __restrict__ xj) {
    __shared__ __align__(16) _Float16 s_rbf[MD * KD];   //  8 KB
    __shared__ __align__(16) _Float16 s_w[FD * KD];     // 16 KB
    __shared__ __align__(16) float    s_g[MD * FD];     // 32 KB
    __shared__ int   s_nbr[MD];
    __shared__ float s_part[FD];

    const int tid = threadIdx.x;
    const int bn  = blockIdx.x;                 // b*N + n

    // rbf tile: 4096 f32 = 1024 float4, convert to f16 in LDS
    const float4* rb = (const float4*)(rbf + (size_t)bn * (MD * KD));
#pragma unroll
    for (int j = 0; j < 4; ++j) {
        int i = tid + j * 256;
        float4 v = rb[i];
        s_rbf[i * 4 + 0] = (_Float16)v.x;
        s_rbf[i * 4 + 1] = (_Float16)v.y;
        s_rbf[i * 4 + 2] = (_Float16)v.z;
        s_rbf[i * 4 + 3] = (_Float16)v.w;
    }
    stage16(k2f_h, s_w, (FD * KD) / 8, tid);    // 1024 chunks
    if (tid < MD) s_nbr[tid] = nbr[bn * MD + tid];
    __syncthreads();

    const int wave = tid >> 5, lane = tid & 31;
    const int mt = wave >> 1;                   // m-tile 0..3
    const int fg = wave & 1;                    // f half
#pragma unroll
    for (int ct = 0; ct < 4; ++ct) {
        const int f0 = (fg * 4 + ct) * 16;
        v8f acc = {};
#pragma unroll
        for (int ks = 0; ks < 2; ++ks) {
            v16h a = load_a_frag(s_rbf, KD, mt * 16, ks * 32, lane);
            v16h b = load_b_frag(s_w,   KD, f0,      ks * 32, lane);
            acc = __builtin_amdgcn_wmma_f32_16x16x32_f16(
                      false, a, false, b, (short)0, acc, false, false);
        }
        const int col = f0 + (lane & 15);
        const int rb2 = mt * 16 + ((lane >> 4) & 1) * 8;
#pragma unroll
        for (int r = 0; r < 8; ++r) s_g[(rb2 + r) * FD + col] = acc[r];
    }
    __syncthreads();

    // gather-reduce: 2 half-blocks over m, 128 f-lanes each
    const int f = tid & 127;
    const int h = tid >> 7;
    const int rowbase = (bn >> 8) << 8;         // b*N
    float acc = 0.0f;
#pragma unroll 4
    for (int m = h * 32; m < h * 32 + 32; ++m) {
        const int nb = s_nbr[m];
        acc += xa[(size_t)(rowbase + nb) * FD + f] * s_g[m * FD + f];
    }
    if (h == 1) s_part[f] = acc;
    __syncthreads();
    if (h == 0) xj[(size_t)bn * FD + f] = acc + s_part[f];
}

// ---- fused 12-layer per-atom MLP chain -------------------------------------
// One block = 64 atom rows; activation state lives in LDS across all layers.
__global__ __launch_bounds__(256) void chain_kernel(
        const float* __restrict__ XA, const float* __restrict__ XJ,
        const float* __restrict__ x,
        const _Float16* __restrict__ h_Wi,  const float* __restrict__ bi,
        const _Float16* __restrict__ h_iW1, const float* __restrict__ ib1,
        const _Float16* __restrict__ h_iW2, const float* __restrict__ ib2,
        const _Float16* __restrict__ h_Wint, const float* __restrict__ bint,
        const float* __restrict__ ug,
        const _Float16* __restrict__ h_aW1, const float* __restrict__ ab1,
        const _Float16* __restrict__ h_aW2, const float* __restrict__ ab2,
        float* __restrict__ out) {
    __shared__ __align__(16) float    sA[64 * FD];      // f32 state, 32 KB
    __shared__ __align__(16) _Float16 sH[64 * FD];      // f16 GEMM input, 16 KB
    __shared__ __align__(16) _Float16 sW[FD * FD];      // f16 weights, 32 KB

    const int tid  = threadIdx.x;
    const int wave = tid >> 5, lane = tid & 31;
    const int row0  = (wave >> 1) * 16;
    const int fbase = (wave & 1) * 64;
    const size_t rowbase = (size_t)blockIdx.x * 64;
    const int WCH = (FD * FD) / 8;              // 2048 16B chunks per weight

    v8f c[4];
    const int colb = fbase + (lane & 15);
    const int rb2  = row0 + ((lane >> 4) & 1) * 8;

    // ---- layer 1: v = ssp(xa @ Wi.T + bi) + xj -----------------------------
    {
        const float* ip = XA + rowbase * FD;
        for (int i = tid; i < 64 * FD; i += 256) sH[i] = (_Float16)ip[i];
        stage16(h_Wi, sW, WCH, tid);
        __syncthreads();
        gemm4(sH, sW, row0, fbase, lane, c);
#pragma unroll
        for (int ct = 0; ct < 4; ++ct) {
            const int col = colb + ct * 16;
#pragma unroll
            for (int r = 0; r < 8; ++r) {
                const int row = rb2 + r;
                const size_t idx = (rowbase + row) * FD + col;
                sA[row * FD + col] = sspf(c[ct][r] + bi[col]) + XJ[idx];
            }
        }
        __syncthreads();
    }

    // ---- 3 interaction residual blocks -------------------------------------
    for (int l = 0; l < 3; ++l) {
        for (int i = tid; i < 64 * FD; i += 256) sH[i] = (_Float16)sspf(sA[i]);
        stage16(h_iW1 + (size_t)l * FD * FD, sW, WCH, tid);
        __syncthreads();
        gemm4(sH, sW, row0, fbase, lane, c);
#pragma unroll
        for (int ct = 0; ct < 4; ++ct) {
            const int col = colb + ct * 16;
#pragma unroll
            for (int r = 0; r < 8; ++r) c[ct][r] = sspf(c[ct][r] + ib1[l * FD + col]);
        }
        __syncthreads();                         // all sH/sW reads done
#pragma unroll
        for (int ct = 0; ct < 4; ++ct) {
            const int col = colb + ct * 16;
#pragma unroll
            for (int r = 0; r < 8; ++r) sH[(rb2 + r) * FD + col] = (_Float16)c[ct][r];
        }
        stage16(h_iW2 + (size_t)l * FD * FD, sW, WCH, tid);
        __syncthreads();
        gemm4(sH, sW, row0, fbase, lane, c);
#pragma unroll
        for (int ct = 0; ct < 4; ++ct) {
            const int col = colb + ct * 16;
#pragma unroll
            for (int r = 0; r < 8; ++r)
                sA[(rb2 + r) * FD + col] += c[ct][r] + ib2[l * FD + col];
        }
        __syncthreads();
    }

    // ---- out = u_gate * x + ssp(v) @ Wint.T + bint -------------------------
    {
        for (int i = tid; i < 64 * FD; i += 256) sH[i] = (_Float16)sspf(sA[i]);
        stage16(h_Wint, sW, WCH, tid);
        __syncthreads();
        gemm4(sH, sW, row0, fbase, lane, c);
#pragma unroll
        for (int ct = 0; ct < 4; ++ct) {
            const int col = colb + ct * 16;
#pragma unroll
            for (int r = 0; r < 8; ++r) {
                const int row = rb2 + r;
                const size_t idx = (rowbase + row) * FD + col;
                sA[row * FD + col] = c[ct][r] + bint[col] + ug[col] * x[idx];
            }
        }
        __syncthreads();
    }

    // ---- 2 atom residual blocks (last layer streams straight to d_out) -----
    for (int l = 0; l < 2; ++l) {
        for (int i = tid; i < 64 * FD; i += 256) sH[i] = (_Float16)sspf(sA[i]);
        stage16(h_aW1 + (size_t)l * FD * FD, sW, WCH, tid);
        __syncthreads();
        gemm4(sH, sW, row0, fbase, lane, c);
#pragma unroll
        for (int ct = 0; ct < 4; ++ct) {
            const int col = colb + ct * 16;
#pragma unroll
            for (int r = 0; r < 8; ++r) c[ct][r] = sspf(c[ct][r] + ab1[l * FD + col]);
        }
        __syncthreads();
#pragma unroll
        for (int ct = 0; ct < 4; ++ct) {
            const int col = colb + ct * 16;
#pragma unroll
            for (int r = 0; r < 8; ++r) sH[(rb2 + r) * FD + col] = (_Float16)c[ct][r];
        }
        stage16(h_aW2 + (size_t)l * FD * FD, sW, WCH, tid);
        __syncthreads();
        gemm4(sH, sW, row0, fbase, lane, c);
#pragma unroll
        for (int ct = 0; ct < 4; ++ct) {
            const int col = colb + ct * 16;
#pragma unroll
            for (int r = 0; r < 8; ++r) {
                const int row = rb2 + r;
                const float v = sA[row * FD + col] + c[ct][r] + ab2[l * FD + col];
                if (l == 1) out[(rowbase + row) * FD + col] = v;
                else        sA[row * FD + col] = v;
            }
        }
        if (l == 0) __syncthreads();
    }
}

// ---------------------------------------------------------------------------
extern "C" void kernel_launch(void* const* d_in, const int* in_sizes, int n_in,
                              void* d_out, int out_size, void* d_ws, size_t ws_size,
                              hipStream_t stream) {
    (void)in_sizes; (void)n_in; (void)out_size; (void)ws_size;

    const float* x    = (const float*)d_in[0];
    const float* rbf  = (const float*)d_in[1];
    const int*   nbr  = (const int*)  d_in[2];
    const float* k2f  = (const float*)d_in[3];
    const float* Wi   = (const float*)d_in[4];
    const float* bi   = (const float*)d_in[5];
    const float* iW1  = (const float*)d_in[6];
    const float* ib1  = (const float*)d_in[7];
    const float* iW2  = (const float*)d_in[8];
    const float* ib2  = (const float*)d_in[9];
    const float* Wint = (const float*)d_in[10];
    const float* bint = (const float*)d_in[11];
    const float* ug   = (const float*)d_in[12];
    const float* aW1  = (const float*)d_in[13];
    const float* ab1  = (const float*)d_in[14];
    const float* aW2  = (const float*)d_in[15];
    const float* ab2  = (const float*)d_in[16];
    float* out = (float*)d_out;

    // workspace layout
    float* XA = (float*)d_ws;                    // ssp(x), 8192x128 f32
    float* XJ = XA + (size_t)ROWS * FD;
    _Float16* WH = (_Float16*)(XJ + (size_t)ROWS * FD);
    _Float16* h_k2f  = WH;
    _Float16* h_Wi   = h_k2f  + FD * KD;
    _Float16* h_iW1  = h_Wi   + FD * FD;
    _Float16* h_iW2  = h_iW1  + 3 * FD * FD;
    _Float16* h_Wint = h_iW2  + 3 * FD * FD;
    _Float16* h_aW1  = h_Wint + FD * FD;
    _Float16* h_aW2  = h_aW1  + 2 * FD * FD;

    const int tot = ROWS * FD;                   // 1,048,576
    ssp_kernel<<<tot / 256, 256, 0, stream>>>(x, XA, tot);

    auto cvt = [&](const float* s, _Float16* d, int n) {
        cvt_h_kernel<<<(n + 255) / 256, 256, 0, stream>>>(s, d, n);
    };
    cvt(k2f,  h_k2f,  FD * KD);
    cvt(Wi,   h_Wi,   FD * FD);
    cvt(iW1,  h_iW1,  3 * FD * FD);
    cvt(iW2,  h_iW2,  3 * FD * FD);
    cvt(Wint, h_Wint, FD * FD);
    cvt(aW1,  h_aW1,  2 * FD * FD);
    cvt(aW2,  h_aW2,  2 * FD * FD);

    // fused RBF-filter GEMM + gather: one block per (b,n)
    rbf_gather_kernel<<<BD * ND, 256, 0, stream>>>(rbf, nbr, XA, h_k2f, XJ);

    // fused 12-layer per-atom MLP chain
    chain_kernel<<<ROWS / 64, 256, 0, stream>>>(
        XA, XJ, x, h_Wi, bi, h_iW1, ib1, h_iW2, ib2,
        h_Wint, bint, ug, h_aW1, ab1, h_aW2, ab2, out);
}
